// Fastformer_MultiHeadAttention_9620726743629
// MI455X (gfx1250) — matrix operand order repaired and checked by reading, not compiled, not run on hardware
//
#include <hip/hip_runtime.h>
#include <hip/hip_bf16.h>
#include <stdint.h>

// ---------------------------------------------------------------------------
// Fastformer MHA for MI455X (gfx1250):
//   bf16 WMMA GEMMs with TDM (tensor_load_to_lds) double-buffered staging
//   + fused fp32 additive-attention chain.
// B=4, S=2048, D_MODEL=1024, H=16, DEPTH=64
// ---------------------------------------------------------------------------

#define BSZ   4
#define SEQ   2048
#define DM    1024
#define MROWS (BSZ * SEQ)          // 8192

typedef __attribute__((ext_vector_type(16))) __bf16 v16bf;
typedef __attribute__((ext_vector_type(8)))  __bf16 bf8v;
typedef __attribute__((ext_vector_type(8)))  float  v8f;
typedef __attribute__((ext_vector_type(4)))  unsigned int u32x4;
typedef __attribute__((ext_vector_type(8)))  int i32x8;
typedef __attribute__((ext_vector_type(4)))  int i32x4;

union FragBF16 { v16bf v; bf8v h[2]; };

__device__ __forceinline__ unsigned short f2bf(float x) {
    unsigned int u = __float_as_uint(x);
    unsigned int r = u + 0x7fffu + ((u >> 16) & 1u);   // round-to-nearest-even
    return (unsigned short)(r >> 16);
}

// --------------------------- fp32 -> bf16 (x4/thread) ----------------------
__global__ __launch_bounds__(256)
void cvt_bf16_kernel(const float* __restrict__ src, unsigned short* __restrict__ dst) {
    size_t i = ((size_t)blockIdx.x * 256 + threadIdx.x) * 4;
    float4 f = *(const float4*)(src + i);
    uint2 o;
    o.x = (unsigned int)f2bf(f.x) | ((unsigned int)f2bf(f.y) << 16);
    o.y = (unsigned int)f2bf(f.z) | ((unsigned int)f2bf(f.w) << 16);
    *(uint2*)(dst + i) = o;
}

// ------------------- fp32 W[K][N] -> bf16 Wt[N][K] (tiled) -----------------
__global__ __launch_bounds__(256)
void wtrans_bf16_kernel(const float* __restrict__ W, unsigned short* __restrict__ Wt) {
    __shared__ unsigned short tile[32][33];
    const int bx = blockIdx.x * 32;            // n tile
    const int by = blockIdx.y * 32;            // k tile
    const int tx = threadIdx.x & 31;
    const int ty = threadIdx.x >> 5;           // 0..7
    #pragma unroll
    for (int i = 0; i < 32; i += 8)
        tile[ty + i][tx] = f2bf(W[(size_t)(by + ty + i) * DM + bx + tx]);
    __syncthreads();
    #pragma unroll
    for (int i = 0; i < 32; i += 8)
        Wt[(size_t)(bx + ty + i) * DM + by + tx] = tile[tx][ty + i];
}

// -------------------- TDM tile load: 128 rows x 32 bf16 --------------------
// Moves one [128][32] bf16 tile (row stride K in memory) into LDS laid out as
// [128][40] shorts: pad_interval=3 (pad after every 16 DWORDs = 64B row),
// pad_amount=3 (insert 4 DWORDs = 8 shorts) -> 80B LDS row stride.
__device__ __forceinline__ void tdm_load_tile(const unsigned short* gtile,
                                              unsigned lds_addr,
                                              unsigned tensor_rows) {
    unsigned long long ga = (unsigned long long)(size_t)gtile;
    u32x4 g0;
    g0.x = 1u;                                       // count=1, user mode
    g0.y = lds_addr;                                 // LDS byte address
    g0.z = (unsigned)(ga & 0xFFFFFFFFu);             // global_addr[31:0]
    g0.w = (unsigned)((ga >> 32) & 0x1FFFFFFu)       // global_addr[56:32]
         | (2u << 30);                               // type=2 ("image")
    i32x8 g1;
    g1[0] = (int)((1u << 16)                         // data_size=1 -> 2 bytes
                | (1u << 20)                         // pad_enable
                | (3u << 22)                         // pad_interval: 16 DWORDs
                | (3u << 25));                       // pad_amount: 4 DWORDs
    g1[1] = (int)((unsigned)DM << 16);               // tensor_dim0[15:0]=1024
    g1[2] = (int)((tensor_rows & 0xFFFFu) << 16);    // tensor_dim1[15:0]
    g1[3] = (int)(((tensor_rows >> 16) & 0xFFFFu)    // tensor_dim1[31:16]
                | (32u << 16));                      // tile_dim0=32
    g1[4] = 128;                                     // tile_dim1=128, tile_dim2=0
    g1[5] = DM;                                      // tensor_dim0_stride=1024
    g1[6] = 0;
    g1[7] = 0;
    i32x4 z4 = {0, 0, 0, 0};
#if __clang_major__ >= 23
    i32x8 z8 = {0, 0, 0, 0, 0, 0, 0, 0};
    __builtin_amdgcn_tensor_load_to_lds(g0, g1, z4, z4, z8, 0);
#else
    __builtin_amdgcn_tensor_load_to_lds(g0, g1, z4, z4, 0);
#endif
}

// ------------------------------ WMMA GEMM ----------------------------------
// C[M][N] = A[M][K](bf16,row-major) * Bt[N][K](bf16) + bias[N]
// block: 256 thr (8 waves). Tile M=128,N=128,BK=32. Wave sub-tile 32x64.
// Tiles staged by the Tensor Data Mover (wave 0 issues descriptors),
// double-buffered; consumers sync via workgroup barrier after tensorcnt wait.
struct GemmSmem {
    unsigned short As[2][128][40];   // 80B rows (16B aligned)
    unsigned short Bs[2][128][40];
};

__global__ __launch_bounds__(256)
void gemm_bf16_wmma(const unsigned short* __restrict__ A,
                    const unsigned short* __restrict__ Bt,
                    const float* __restrict__ bias,
                    float* __restrict__ C,
                    int N, int K)
{
    __shared__ __align__(16) GemmSmem smem;

    const int t  = threadIdx.x;
    const int m0 = blockIdx.y * 128;
    const int n0 = blockIdx.x * 128;

    const int wave = t >> 5;
    const int lane = t & 31;
    const int wm = (wave & 3) * 32;      // 4 waves along M
    const int wn = (wave >> 2) * 64;     // 2 waves along N
    const int hi = lane >> 4;            // half-wave select
    const int ln = lane & 15;
    const int aoff = hi * 8;             // A frag: K {0-7,16-23} vs {8-15,24-31}
    const int boff = hi * 8;             // B frag: K {0-7,16-23} vs {8-15,24-31}

    const unsigned short* Abase = A  + (size_t)m0 * K;
    const unsigned short* Bbase = Bt + (size_t)n0 * K;
    const unsigned ldsA0 = (unsigned)(size_t)&smem.As[0][0][0];
    const unsigned ldsA1 = (unsigned)(size_t)&smem.As[1][0][0];
    const unsigned ldsB0 = (unsigned)(size_t)&smem.Bs[0][0][0];
    const unsigned ldsB1 = (unsigned)(size_t)&smem.Bs[1][0][0];

    v8f acc[2][4];
    const v8f vzero = {0.f, 0.f, 0.f, 0.f, 0.f, 0.f, 0.f, 0.f};
    #pragma unroll
    for (int r = 0; r < 2; ++r)
        #pragma unroll
        for (int c = 0; c < 4; ++c) acc[r][c] = vzero;

    // prologue: TDM fills buffer 0
    if (wave == 0) {
        tdm_load_tile(Abase, ldsA0, (unsigned)MROWS);
        tdm_load_tile(Bbase, ldsB0, (unsigned)DM);
        __builtin_amdgcn_s_wait_tensorcnt(0);
    }
    __syncthreads();

    const int nk = K >> 5;               // 32 K-steps
    for (int ks = 0; ks < nk; ++ks) {
        const int buf = ks & 1;
        if (wave == 0 && ks + 1 < nk) {  // async-prefetch next tiles via TDM
            const int k1 = (ks + 1) << 5;
            tdm_load_tile(Abase + k1, buf ? ldsA0 : ldsA1, (unsigned)MROWS);
            tdm_load_tile(Bbase + k1, buf ? ldsB0 : ldsB1, (unsigned)DM);
        }

        FragBF16 af[2], bfm[4];
        #pragma unroll
        for (int r = 0; r < 2; ++r) {
            const unsigned short* p = &smem.As[buf][wm + r * 16 + ln][0];
            af[r].h[0] = *(const bf8v*)(p + aoff);
            af[r].h[1] = *(const bf8v*)(p + aoff + 16);
        }
        #pragma unroll
        for (int c = 0; c < 4; ++c) {
            const unsigned short* p = &smem.Bs[buf][wn + c * 16 + ln][0];
            bfm[c].h[0] = *(const bf8v*)(p + boff);
            bfm[c].h[1] = *(const bf8v*)(p + boff + 16);
        }
        #pragma unroll
        for (int r = 0; r < 2; ++r)
            #pragma unroll
            for (int c = 0; c < 4; ++c)
                acc[r][c] = __builtin_amdgcn_wmma_f32_16x16x32_bf16(
                    false, af[r].v, false, bfm[c].v,
                    (short)0, acc[r][c], false, false);

        if (wave == 0 && ks + 1 < nk)
            __builtin_amdgcn_s_wait_tensorcnt(0);  // next tiles resident
        __syncthreads();
    }

    // epilogue: C/D layout -> VGPR e holds M = e + 8*hi, lane&15 = N
    #pragma unroll
    for (int r = 0; r < 2; ++r) {
        #pragma unroll
        for (int c = 0; c < 4; ++c) {
            const int n = n0 + wn + c * 16 + ln;
            const float bb = bias[n];
            #pragma unroll
            for (int e = 0; e < 8; ++e) {
                const int m = m0 + wm + r * 16 + hi * 8 + e;
                C[(size_t)m * N + n] = acc[r][c][e] + bb;
            }
        }
    }
}

// ---------------- weighted column sum: out[b,d] += Σ_s X[b,s,d]*g[s] -------
__global__ __launch_bounds__(256)
void wcolsum_kernel(const float* __restrict__ X, const float* __restrict__ g,
                    float* __restrict__ out) {
    const int d  = blockIdx.x * 256 + threadIdx.x;   // grid.x = DM/256
    const int b  = blockIdx.y;                       // grid.y = B
    const int s0 = blockIdx.z * 128;                 // grid.z = SEQ/128
    const float* xp = X + ((size_t)b * SEQ + s0) * DM + d;
    float acc = 0.f;
    #pragma unroll 4
    for (int s = 0; s < 128; ++s)
        acc += xp[(size_t)s * DM] * g[s0 + s];
    atomicAdd(&out[b * DM + d], acc);
}

// -------- fused fastformer elementwise: obf = bf16(V*Sq*Sk + Q*gqw) --------
__global__ __launch_bounds__(256)
void ffew_kernel(const float* __restrict__ Qf, const float* __restrict__ Vf,
                 const float* __restrict__ Sq, const float* __restrict__ Sk,
                 const float* __restrict__ gqw, unsigned short* __restrict__ obf) {
    const int b = blockIdx.y;
    const int i = blockIdx.x * 256 + threadIdx.x;    // over SEQ*DM
    const int s = i >> 10;
    const int d = i & (DM - 1);
    const size_t idx = (size_t)b * SEQ * DM + i;
    const float gk = Sq[b * DM + d] * Sk[b * DM + d];
    obf[idx] = f2bf(Vf[idx] * gk + Qf[idx] * gqw[s]);
}

// ---------------------------------------------------------------------------
extern "C" void kernel_launch(void* const* d_in, const int* in_sizes, int n_in,
                              void* d_out, int out_size, void* d_ws, size_t ws_size,
                              hipStream_t stream) {
    (void)in_sizes; (void)n_in; (void)out_size; (void)ws_size;
    const float* v   = (const float*)d_in[0];
    const float* k   = (const float*)d_in[1];
    const float* q   = (const float*)d_in[2];
    // d_in[3] mask: unused by the layer
    const float* wq  = (const float*)d_in[4];
    const float* bq  = (const float*)d_in[5];
    const float* wk  = (const float*)d_in[6];
    const float* bk  = (const float*)d_in[7];
    const float* wv  = (const float*)d_in[8];
    const float* bv  = (const float*)d_in[9];
    const float* gqw = (const float*)d_in[10];
    const float* gkw = (const float*)d_in[11];
    const float* wo  = (const float*)d_in[12];
    const float* bo  = (const float*)d_in[13];
    float* out = (float*)d_out;

    // workspace carve (256B aligned regions)
    char* wsp = (char*)d_ws;
    auto carve = [&](size_t bytes) -> void* {
        void* p = wsp;
        wsp += (bytes + 255) & ~(size_t)255;
        return p;
    };
    const size_t ACT   = (size_t)MROWS * DM;         // 8.4M elements
    unsigned short* qbf = (unsigned short*)carve(ACT * 2);
    unsigned short* kbf = (unsigned short*)carve(ACT * 2);
    unsigned short* vbf = (unsigned short*)carve(ACT * 2);
    unsigned short* wqT = (unsigned short*)carve((size_t)DM * DM * 2);
    unsigned short* wkT = (unsigned short*)carve((size_t)DM * DM * 2);
    unsigned short* wvT = (unsigned short*)carve((size_t)DM * DM * 2);
    unsigned short* woT = (unsigned short*)carve((size_t)DM * DM * 2);
    float* Qf = (float*)carve(ACT * 4);
    float* Kf = (float*)carve(ACT * 4);
    float* Vf = (float*)carve(ACT * 4);
    float* Sq = (float*)carve((size_t)BSZ * DM * 4);
    float* Sk = (float*)carve((size_t)BSZ * DM * 4);
    unsigned short* obf = (unsigned short*)carve(ACT * 2);

    // 1) activation fp32 -> bf16
    const int cvtBlocks = (int)(ACT / 4 / 256);      // 8192
    cvt_bf16_kernel<<<cvtBlocks, 256, 0, stream>>>(q, qbf);
    cvt_bf16_kernel<<<cvtBlocks, 256, 0, stream>>>(k, kbf);
    cvt_bf16_kernel<<<cvtBlocks, 256, 0, stream>>>(v, vbf);

    // 2) weight fp32 [K][N] -> bf16 transposed [N][K]
    dim3 tgrid(DM / 32, DM / 32);
    wtrans_bf16_kernel<<<tgrid, 256, 0, stream>>>(wq, wqT);
    wtrans_bf16_kernel<<<tgrid, 256, 0, stream>>>(wk, wkT);
    wtrans_bf16_kernel<<<tgrid, 256, 0, stream>>>(wv, wvT);
    wtrans_bf16_kernel<<<tgrid, 256, 0, stream>>>(wo, woT);

    // 3) QKV projection GEMMs (bf16 WMMA, f32 accumulate, +bias)
    dim3 ggrid(DM / 128, MROWS / 128);               // (8, 64)
    gemm_bf16_wmma<<<ggrid, 256, 0, stream>>>(qbf, wqT, bq, Qf, DM, DM);
    gemm_bf16_wmma<<<ggrid, 256, 0, stream>>>(kbf, wkT, bk, Kf, DM, DM);
    gemm_bf16_wmma<<<ggrid, 256, 0, stream>>>(vbf, wvT, bv, Vf, DM, DM);

    // 4) zero Sq/Sk (contiguous in ws), then weighted column sums
    hipMemsetAsync(Sq, 0, 2 * (size_t)BSZ * DM * sizeof(float), stream);
    dim3 cgrid(DM / 256, BSZ, SEQ / 128);            // (4, 4, 16)
    wcolsum_kernel<<<cgrid, 256, 0, stream>>>(Qf, gqw, Sq);
    wcolsum_kernel<<<cgrid, 256, 0, stream>>>(Kf, gkw, Sk);

    // 5) fused fastformer elementwise -> bf16 input of output projection
    dim3 egrid(SEQ * DM / 256, BSZ);                 // (8192, 4)
    ffew_kernel<<<egrid, 256, 0, stream>>>(Qf, Vf, Sq, Sk, gqw, obf);

    // 6) output projection GEMM -> d_out (fp32)
    gemm_bf16_wmma<<<ggrid, 256, 0, stream>>>(obf, woT, bo, out, DM, DM);
}
